// LAT_16260746182953
// MI455X (gfx1250) — matrix operand order repaired and checked by reading, not compiled, not run on hardware
//
#include <hip/hip_runtime.h>
#include <hip/hip_bf16.h>

typedef __bf16 bf16_t;
typedef __attribute__((ext_vector_type(16))) __bf16 v16bf;
typedef __attribute__((ext_vector_type(8)))  __bf16 v8bf;
typedef __attribute__((ext_vector_type(8)))  float  v8f;
typedef __attribute__((ext_vector_type(4)))  float  f32x4;

#define GEMM_BM 128
#define GEMM_BN 128
#define GEMM_BK 32
#define LDS_K   56   // padded bf16 K-stride: 112B = 16B-aligned, bank-conflict-free row fan

// ---------------------------------------------------------------------------
// WMMA bf16 GEMM:  C[M,N] = A[M,K] @ W[N,K]^T   (+bias, +relu when ksplit==1)
// Block tile 128x128, 8 waves as 4M x 2N, wave tile 32x64 (2x4 WMMA tiles).
// Branch-free inner loop (row-clamped loads), register-staged global->LDS
// pipeline.  ksplit>1: grid.z splits K, partials accumulated with global
// f32 atomics into pre-zeroed C.  K and K/ksplit must be multiples of 32.
// ---------------------------------------------------------------------------
__global__ __launch_bounds__(256)
void wmma_gemm_kernel(const float* __restrict__ A, const float* __restrict__ W,
                      const float* __restrict__ bias, float* __restrict__ C,
                      int M, int N, int K, int relu, int ksplit)
{
    __shared__ __align__(16) bf16_t sA[GEMM_BM * LDS_K];
    __shared__ __align__(16) bf16_t sB[GEMM_BN * LDS_K];

    const int tid  = threadIdx.x;
    const int lane = tid & 31;
    const int wv   = tid >> 5;      // 0..7
    const int wm   = wv & 3;        // 4 M-waves * 32 rows
    const int wn   = wv >> 2;       // 2 N-waves * 64 cols
    const int half = lane >> 4;     // 0 | 1
    const int l16  = lane & 15;

    const int m0 = blockIdx.y * GEMM_BM;
    const int n0 = blockIdx.x * GEMM_BN;
    const int kchunk = K / ksplit;
    const int kbeg = blockIdx.z * kchunk;
    const int kend = kbeg + kchunk;

    // staging: 128x32 panel = 4096 elems; 256 threads * 2 segments of 8
    const int seg0  = tid * 2;
    const int row0  = seg0 >> 2;
    const int ksg0  = (seg0 & 3) * 8;
    const int row1  = (seg0 + 1) >> 2;
    const int ksg1  = ((seg0 + 1) & 3) * 8;

    // clamped source rows: loads always legal, bogus rows never stored
    const int gmA0 = min(m0 + row0, M - 1);
    const int gmA1 = min(m0 + row1, M - 1);
    const int gnB0 = min(n0 + row0, N - 1);
    const int gnB1 = min(n0 + row1, N - 1);

    f32x4 a0lo, a0hi, a1lo, a1hi, b0lo, b0hi, b1lo, b1hi;

    auto load_stage = [&](int kt) {
        const float* pa0 = A + (size_t)gmA0 * K + kt + ksg0;
        const float* pa1 = A + (size_t)gmA1 * K + kt + ksg1;
        const float* pb0 = W + (size_t)gnB0 * K + kt + ksg0;
        const float* pb1 = W + (size_t)gnB1 * K + kt + ksg1;
        a0lo = *(const f32x4*)(pa0);  a0hi = *(const f32x4*)(pa0 + 4);
        a1lo = *(const f32x4*)(pa1);  a1hi = *(const f32x4*)(pa1 + 4);
        b0lo = *(const f32x4*)(pb0);  b0hi = *(const f32x4*)(pb0 + 4);
        b1lo = *(const f32x4*)(pb1);  b1hi = *(const f32x4*)(pb1 + 4);
    };

    auto store_stage = [&]() {
        bf16_t* da0 = &sA[row0 * LDS_K + ksg0];
        bf16_t* da1 = &sA[row1 * LDS_K + ksg1];
        bf16_t* db0 = &sB[row0 * LDS_K + ksg0];
        bf16_t* db1 = &sB[row1 * LDS_K + ksg1];
        da0[0]=(bf16_t)a0lo.x; da0[1]=(bf16_t)a0lo.y; da0[2]=(bf16_t)a0lo.z; da0[3]=(bf16_t)a0lo.w;
        da0[4]=(bf16_t)a0hi.x; da0[5]=(bf16_t)a0hi.y; da0[6]=(bf16_t)a0hi.z; da0[7]=(bf16_t)a0hi.w;
        da1[0]=(bf16_t)a1lo.x; da1[1]=(bf16_t)a1lo.y; da1[2]=(bf16_t)a1lo.z; da1[3]=(bf16_t)a1lo.w;
        da1[4]=(bf16_t)a1hi.x; da1[5]=(bf16_t)a1hi.y; da1[6]=(bf16_t)a1hi.z; da1[7]=(bf16_t)a1hi.w;
        db0[0]=(bf16_t)b0lo.x; db0[1]=(bf16_t)b0lo.y; db0[2]=(bf16_t)b0lo.z; db0[3]=(bf16_t)b0lo.w;
        db0[4]=(bf16_t)b0hi.x; db0[5]=(bf16_t)b0hi.y; db0[6]=(bf16_t)b0hi.z; db0[7]=(bf16_t)b0hi.w;
        db1[0]=(bf16_t)b1lo.x; db1[1]=(bf16_t)b1lo.y; db1[2]=(bf16_t)b1lo.z; db1[3]=(bf16_t)b1lo.w;
        db1[4]=(bf16_t)b1hi.x; db1[5]=(bf16_t)b1hi.y; db1[6]=(bf16_t)b1hi.z; db1[7]=(bf16_t)b1hi.w;
    };

    v8f acc[2][4] = {};

    load_stage(kbeg);
    for (int kt = kbeg; kt < kend; kt += GEMM_BK) {
        store_stage();
        __syncthreads();
        if (kt + GEMM_BK < kend) load_stage(kt + GEMM_BK);  // overlap with WMMA below

        // ---- A fragments (ISA 16-bit A 16x32 layout) ----
        v16bf afr[2];
        #pragma unroll
        for (int mt = 0; mt < 2; ++mt) {
            const bf16_t* ar = &sA[(wm * 32 + mt * 16 + l16) * LDS_K];
            v8bf lo = *(const v8bf*)(ar + 8 * half);        // e=0..7  : k = e + 8*half
            v8bf hi = *(const v8bf*)(ar + 16 + 8 * half);   // e=8..15 : k = 16+(e-8)+8*half
            afr[mt] = __builtin_shufflevector(lo, hi,
                0,1,2,3,4,5,6,7,8,9,10,11,12,13,14,15);
        }
        // ---- B fragments (row = column n, same k order as A) ----
        v16bf bfr[4];
        #pragma unroll
        for (int nt = 0; nt < 4; ++nt) {
            const bf16_t* br = &sB[(wn * 64 + nt * 16 + l16) * LDS_K];
            v8bf lo = *(const v8bf*)(br + 8 * half);
            v8bf hi = *(const v8bf*)(br + 16 + 8 * half);
            bfr[nt] = __builtin_shufflevector(lo, hi,
                0,1,2,3,4,5,6,7,8,9,10,11,12,13,14,15);
        }
        // ---- 8 WMMAs back-to-back ----
        #pragma unroll
        for (int mt = 0; mt < 2; ++mt)
            #pragma unroll
            for (int nt = 0; nt < 4; ++nt)
                acc[mt][nt] = __builtin_amdgcn_wmma_f32_16x16x32_bf16(
                    false, afr[mt], false, bfr[nt], (short)0, acc[mt][nt], false, false);
        __syncthreads();
    }

    // ---- epilogue: C/D layout VGPR r -> M = r + 8*half, N = l16 ----
    #pragma unroll
    for (int mt = 0; mt < 2; ++mt) {
        #pragma unroll
        for (int nt = 0; nt < 4; ++nt) {
            #pragma unroll
            for (int r = 0; r < 8; ++r) {
                const int m = m0 + wm * 32 + mt * 16 + r + 8 * half;
                const int n = n0 + wn * 64 + nt * 16 + l16;
                if (m < M && n < N) {
                    if (ksplit > 1) {
                        atomicAdd(&C[(size_t)m * N + n], acc[mt][nt][r]);
                    } else {
                        float v = acc[mt][nt][r];
                        if (bias) v += bias[n];
                        if (relu) v = fmaxf(v, 0.0f);
                        C[(size_t)m * N + n] = v;
                    }
                }
            }
        }
    }
}

// ---------------------------------------------------------------------------
// per-row: relu(x + bias) -> LayerNorm(g, beta), permute rows
// [B,V,J]-order -> [V, B*J]-order for attention.  One wave per row of 512.
// ---------------------------------------------------------------------------
__global__ __launch_bounds__(256)
void bias_relu_ln_permute(const float* __restrict__ in, const float* __restrict__ bias,
                          const float* __restrict__ g, const float* __restrict__ beta,
                          float* __restrict__ out)
{
    const int gw   = (blockIdx.x * blockDim.x + threadIdx.x) >> 5;
    const int lane = threadIdx.x & 31;
    if (gw >= 672) return;

    const float* x = in + (size_t)gw * 512;
    float v[16];
    float s = 0.0f;
    #pragma unroll
    for (int i = 0; i < 16; ++i) {
        const int e = lane + 32 * i;
        float t = x[e] + bias[e];
        t = fmaxf(t, 0.0f);
        v[i] = t;
        s += t;
    }
    #pragma unroll
    for (int off = 16; off > 0; off >>= 1) s += __shfl_xor(s, off, 32);
    const float mu = s * (1.0f / 512.0f);
    float q = 0.0f;
    #pragma unroll
    for (int i = 0; i < 16; ++i) { const float d = v[i] - mu; q += d * d; }
    #pragma unroll
    for (int off = 16; off > 0; off >>= 1) q += __shfl_xor(q, off, 32);
    const float inv = rsqrtf(q * (1.0f / 512.0f) + 1e-5f);

    const int j  = gw % 21;
    const int bv = gw / 21;
    const int vw = bv & 3;     // view
    const int b  = bv >> 2;    // batch
    float* o = out + ((size_t)(vw * 168 + b * 21 + j)) * 512;
    #pragma unroll
    for (int i = 0; i < 16; ++i) {
        const int e = lane + 32 * i;
        o[e] = (v[i] - mu) * inv * g[e] + beta[e];
    }
}

// ---------------------------------------------------------------------------
// MHA core: qkv [4*168, 1536] rows = s*168+n.  One wave per (n, head);
// hd=64 -> 2 floats/lane.  Writes o_att [4*168, 512].
// ---------------------------------------------------------------------------
__global__ __launch_bounds__(256)
void mha_core(const float* __restrict__ qkv, float* __restrict__ out)
{
    const int gw   = (blockIdx.x * blockDim.x + threadIdx.x) >> 5;
    const int lane = threadIdx.x & 31;
    if (gw >= 168 * 8) return;
    const int n = gw >> 3;
    const int h = gw & 7;

    float q[4][2], k[4][2], v[4][2];
    #pragma unroll
    for (int s = 0; s < 4; ++s) {
        const float* r = qkv + ((size_t)(s * 168 + n)) * 1536 + h * 64;
        q[s][0] = r[lane];        q[s][1] = r[32 + lane];
        k[s][0] = r[512 + lane];  k[s][1] = r[544 + lane];
        v[s][0] = r[1024 + lane]; v[s][1] = r[1056 + lane];
    }
    float sc[4][4];
    #pragma unroll
    for (int si = 0; si < 4; ++si)
        #pragma unroll
        for (int sj = 0; sj < 4; ++sj) {
            float p = q[si][0] * k[sj][0] + q[si][1] * k[sj][1];
            #pragma unroll
            for (int off = 16; off > 0; off >>= 1) p += __shfl_xor(p, off, 32);
            sc[si][sj] = p * 0.125f;   // 1/sqrt(64)
        }
    #pragma unroll
    for (int si = 0; si < 4; ++si) {
        const float mx = fmaxf(fmaxf(sc[si][0], sc[si][1]), fmaxf(sc[si][2], sc[si][3]));
        const float e0 = __expf(sc[si][0] - mx);
        const float e1 = __expf(sc[si][1] - mx);
        const float e2 = __expf(sc[si][2] - mx);
        const float e3 = __expf(sc[si][3] - mx);
        const float inv = 1.0f / (e0 + e1 + e2 + e3);
        const float o0 = (e0 * v[0][0] + e1 * v[1][0] + e2 * v[2][0] + e3 * v[3][0]) * inv;
        const float o1 = (e0 * v[0][1] + e1 * v[1][1] + e2 * v[2][1] + e3 * v[3][1]) * inv;
        float* orow = out + ((size_t)(si * 168 + n)) * 512 + h * 64;
        orow[lane]      = o0;
        orow[32 + lane] = o1;
    }
}

// mean over S=4 of [4*168, 512] -> out [168, 512]
__global__ __launch_bounds__(256)
void mean_over_s(const float* __restrict__ in, float* __restrict__ out)
{
    const int i = blockIdx.x * blockDim.x + threadIdx.x;
    if (i >= 168 * 512) return;
    const int n = i >> 9;
    const int e = i & 511;
    float s = 0.0f;
    #pragma unroll
    for (int sq = 0; sq < 4; ++sq) s += in[((size_t)(sq * 168 + n)) * 512 + e];
    out[i] = 0.25f * s;
}

// final: out[n,c] = joints[n,c] + r1[n,:] @ W2[c,:] + b2[c]; one wave per row
__global__ __launch_bounds__(256)
void refine_out(const float* __restrict__ r1, const float* __restrict__ W2,
                const float* __restrict__ b2, const float* __restrict__ joints,
                float* __restrict__ out)
{
    const int gw   = (blockIdx.x * blockDim.x + threadIdx.x) >> 5;
    const int lane = threadIdx.x & 31;
    if (gw >= 168) return;
    const float* x = r1 + (size_t)gw * 256;
    float xs[8];
    #pragma unroll
    for (int i = 0; i < 8; ++i) xs[i] = x[lane + 32 * i];
    #pragma unroll
    for (int c = 0; c < 3; ++c) {
        float p = 0.0f;
        #pragma unroll
        for (int i = 0; i < 8; ++i) p += xs[i] * W2[c * 256 + lane + 32 * i];
        #pragma unroll
        for (int off = 16; off > 0; off >>= 1) p += __shfl_xor(p, off, 32);
        if (lane == 0) out[gw * 3 + c] = joints[gw * 3 + c] + p + b2[c];
    }
}

// ---------------------------------------------------------------------------
extern "C" void kernel_launch(void* const* d_in, const int* in_sizes, int n_in,
                              void* d_out, int out_size, void* d_ws, size_t ws_size,
                              hipStream_t stream) {
    (void)in_sizes; (void)n_in; (void)out_size; (void)ws_size;

    const float* jaf[4] = { (const float*)d_in[0], (const float*)d_in[1],
                            (const float*)d_in[2], (const float*)d_in[3] };
    const float* joints = (const float*)d_in[4];
    const float* lvlW[4], *lvlB[4], *lvlG[4], *lvlBeta[4];
    for (int l = 0; l < 4; ++l) {
        lvlW[l]    = (const float*)d_in[5 + 4 * l];
        lvlB[l]    = (const float*)d_in[6 + 4 * l];
        lvlG[l]    = (const float*)d_in[7 + 4 * l];
        lvlBeta[l] = (const float*)d_in[8 + 4 * l];
    }
    const float* vWi = (const float*)d_in[21]; const float* vbi = (const float*)d_in[22];
    const float* vWo = (const float*)d_in[23]; const float* vbo = (const float*)d_in[24];
    const float* lWi = (const float*)d_in[25]; const float* lbi = (const float*)d_in[26];
    const float* lWo = (const float*)d_in[27]; const float* lbo = (const float*)d_in[28];
    const float* rW1 = (const float*)d_in[29]; const float* rb1 = (const float*)d_in[30];
    const float* rW2 = (const float*)d_in[31]; const float* rb2 = (const float*)d_in[32];

    // workspace carve-up (floats)
    float* ws    = (float*)d_ws;
    float* pre   = ws;                 // [672, 512]
    float* hbuf  = pre   + 344064;     // [672, 512]  (V-major rows)
    float* qkv   = hbuf  + 344064;     // [672, 1536]
    float* attnb = qkv   + 1032192;    // [672, 512]
    float* proj  = attnb + 344064;     // [672, 512]
    float* Lbuf  = proj  + 344064;     // [4*168, 512]
    float* fused = Lbuf  + 344064;     // [168, 512]
    float* r1buf = fused + 86016;      // [168, 256]

    const int  Klvl[4]   = { 256 * 49, 512 * 49, 1024 * 49, 2048 * 49 };
    const int  ksplit[4] = { 8, 16, 16, 32 };   // K/ksplit stays a multiple of 32

    auto gemm = [&](const float* A, const float* W, const float* bias, float* C,
                    int M, int N, int K, int relu, int ks) {
        dim3 grid((N + GEMM_BN - 1) / GEMM_BN, (M + GEMM_BM - 1) / GEMM_BM, ks);
        hipLaunchKernelGGL(wmma_gemm_kernel, grid, dim3(256), 0, stream,
                           A, W, bias, C, M, N, K, relu, ks);
    };

    // per-level: projection -> LN -> view attention -> mean over views
    for (int l = 0; l < 4; ++l) {
        hipMemsetAsync(pre, 0, (size_t)344064 * sizeof(float), stream);
        gemm(jaf[l], lvlW[l], nullptr, pre, 672, 512, Klvl[l], 0, ksplit[l]);
        hipLaunchKernelGGL(bias_relu_ln_permute, dim3(84), dim3(256), 0, stream,
                           pre, lvlB[l], lvlG[l], lvlBeta[l], hbuf);
        gemm(hbuf, vWi, vbi, qkv, 672, 1536, 512, 0, 1);
        hipLaunchKernelGGL(mha_core, dim3(168), dim3(256), 0, stream, qkv, attnb);
        gemm(attnb, vWo, vbo, proj, 672, 512, 512, 0, 1);
        hipLaunchKernelGGL(mean_over_s, dim3((168 * 512 + 255) / 256), dim3(256), 0, stream,
                           proj, Lbuf + (size_t)l * 168 * 512);
    }

    // level attention -> mean over levels
    gemm(Lbuf, lWi, lbi, qkv, 672, 1536, 512, 0, 1);
    hipLaunchKernelGGL(mha_core, dim3(168), dim3(256), 0, stream, qkv, attnb);
    gemm(attnb, lWo, lbo, proj, 672, 512, 512, 0, 1);
    hipLaunchKernelGGL(mean_over_s, dim3((168 * 512 + 255) / 256), dim3(256), 0, stream,
                       proj, fused);

    // refine MLP + residual joints
    gemm(fused, rW1, rb1, r1buf, 168, 256, 512, 1, 1);
    hipLaunchKernelGGL(refine_out, dim3(21), dim3(256), 0, stream,
                       r1buf, rW2, rb2, joints, (float*)d_out);
}
